// SynBlock_75522704933204
// MI455X (gfx1250) — matrix operand order repaired and checked by reading, not compiled, not run on hardware
//
#include <hip/hip_runtime.h>
#include <stdint.h>

// ---------------------------------------------------------------------------
// AdaCoF SynBlock for MI455X (gfx1250).
// All dense convs: implicit GEMM on v_wmma_f32_16x16x32_f16 with
//  - tap-major, zero-padded f16 weights  [CoutPad][taps*CPAD]
//  - NHWC f16 activations with zeroed 1-px halo and zero-padded channels
//  - 16(cout) x 32(pixels) tiles per wave: A (weights) shared by 2 wmmas
// ConvT uses [tap][Cout][CinPad] repacked f32 weights so both streams are
// channel-contiguous and vectorize.
// ---------------------------------------------------------------------------

typedef _Float16 half_t;
typedef __attribute__((ext_vector_type(16))) _Float16 v16h;
typedef __attribute__((ext_vector_type(8)))  _Float16 v8h;
typedef __attribute__((ext_vector_type(8)))  float    v8f;

#define B_  2
#define C_  64
#define T_  4
#define HL  128
#define WL  224
#define HH  256
#define WH  448
#define F2_ 25
#define SL_ (HL * WL)    // 28672
#define SH_ (HH * WH)    // 114688

__device__ __forceinline__ int imin(int a, int b) { return a < b ? a : b; }
__device__ __forceinline__ int imax(int a, int b) { return a > b ? a : b; }

#define SHUF16(lo, hi) __builtin_shufflevector(lo, hi, 0,1,2,3,4,5,6,7,8,9,10,11,12,13,14,15)

// ---------------------------------------------------------------------------
// Elementwise helpers
// ---------------------------------------------------------------------------
__global__ void zero_u32(uint32_t* __restrict__ p, long long n) {
  long long i = (long long)blockIdx.x * blockDim.x + threadIdx.x;
  if (i < n) p[i] = 0u;
}
__global__ void zero_f32(float* __restrict__ p, int n) {
  int i = blockIdx.x * blockDim.x + threadIdx.x;
  if (i < n) p[i] = 0.f;
}

// Repack torch conv weight [Cout][Cin][taps] into tap-major zero-padded f16:
// out[co][tap*CPAD + ci], co < CoutPad, ci < CPAD.
__global__ void repack_w(const float* __restrict__ w, half_t* __restrict__ out,
                         int Cout, int Cin, int CoutPad, int CPAD, int TAPS) {
  int i = blockIdx.x * blockDim.x + threadIdx.x;
  int total = CoutPad * TAPS * CPAD;
  if (i >= total) return;
  int ci  = i % CPAD;
  int t2  = i / CPAD;
  int tap = t2 % TAPS;
  int co  = t2 / TAPS;
  half_t v = (half_t)0.f;
  if (co < Cout && ci < Cin) v = (half_t)w[((size_t)co * Cin + ci) * TAPS + tap];
  out[i] = v;
}

// Repack convT weight [Cin][Cout][9] -> [9][Cout][CinP] f32, zero-padded.
__global__ void repack_wt(const float* __restrict__ w, float* __restrict__ out,
                          int Cin, int Cout, int CinP) {
  int i = blockIdx.x * blockDim.x + threadIdx.x;
  int total = 9 * Cout * CinP;
  if (i >= total) return;
  int c   = i % CinP;
  int t2  = i / CinP;
  int o   = t2 % Cout;
  int tap = t2 / Cout;
  out[i] = (c < Cin) ? w[((size_t)c * Cout + o) * 9 + tap] : 0.f;
}

// fea [B,C,T,hl,wl] -> NHWC halo f16 [B,(HL+2),(WL+2),64] interior (fixed t).
__global__ void feat_tr(const float* __restrict__ fea, half_t* __restrict__ out, int t) {
  __shared__ _Float16 tile[64][65];
  const int PT = SL_ / 64;                       // 448 pixel tiles per image
  int blk = blockIdx.x;
  int b  = blk / PT;
  int p0 = (blk % PT) * 64;
  int tx = threadIdx.x & 63;
  int ty = threadIdx.x >> 6;                     // 0..3
  for (int c = ty; c < 64; c += 4)
    tile[c][tx] = (half_t)fea[(((size_t)b * 64 + c) * T_ + t) * SL_ + p0 + tx];
  __syncthreads();
  for (int pp = ty; pp < 64; pp += 4) {
    int p = p0 + pp;
    int y = p / WL, x = p % WL;
    out[(((size_t)b * (HL + 2) + y + 1) * (WL + 2) + x + 1) * 64 + tx] = tile[tx][pp];
  }
}

// fea [B,256,SL] (C*T flattened) -> pixel-major NHWC f16 [B][SL][256].
__global__ void fuse_tr(const float* __restrict__ fea, half_t* __restrict__ out) {
  __shared__ _Float16 tile[64][65];
  const int PT = SL_ / 64;                       // 448
  int blk = blockIdx.x;
  int ct = blk % 4;  blk /= 4;                   // channel tile (4 x 64 = 256)
  int pt = blk % PT;
  int b  = blk / PT;
  int c0 = ct * 64, p0 = pt * 64;
  int tx = threadIdx.x & 63;
  int ty = threadIdx.x >> 6;
  for (int c = ty; c < 64; c += 4)
    tile[c][tx] = (half_t)fea[((size_t)b * 256 + c0 + c) * SL_ + p0 + tx];
  __syncthreads();
  for (int pp = ty; pp < 64; pp += 4)
    out[((size_t)b * SL_ + p0 + pp) * 256 + c0 + tx] = tile[tx][pp];
}

// ---------------------------------------------------------------------------
// 3x3 conv, implicit GEMM on WMMA; 16x32 output tile per wave (A shared).
//   x   : [N][(H+2)][(W+2)][CPAD] f16, halo+pad zeroed
//   w16 : [COUTP][9*CPAD] f16 tap-major
//   out : OUT16 ? NHWC f16 (OHALO: with halo) : NCHW f32 flat
// ---------------------------------------------------------------------------
template<int H, int W, int CPAD, int COUTP, int COUT, bool LRELU, bool OUT16, bool OHALO>
__global__ void conv3_wmma(const half_t* __restrict__ x,
                           const half_t* __restrict__ w16,
                           const float*  __restrict__ bias,
                           void*         __restrict__ outp, int N) {
  constexpr int WT = W / 32;                     // 32-pixel tiles (W%32==0)
  constexpr int CT = COUTP / 16;
  constexpr int Ws = W + 2;
  int wave = (blockIdx.x * blockDim.x + threadIdx.x) >> 5;
  int lane = threadIdx.x & 31;
  if (wave >= N * H * WT * CT) return;

  int mt = wave % CT; int pt = wave / CT;
  int xt = pt % WT;  pt /= WT;
  int y  = pt % H;
  int n  = pt / H;

  const int hs  = lane >> 4;
  const int l15 = lane & 15;
  const int kA  = hs * 8;
  const int kB  = hs * 16;
  const int px  = xt * 32 + l15;                 // first pixel sub-tile

  const half_t* xn = x + (size_t)n * (H + 2) * Ws * CPAD;
  const half_t* wr = w16 + (size_t)(mt * 16 + l15) * (9 * CPAD);

  v8f acc0 = {0.f, 0.f, 0.f, 0.f, 0.f, 0.f, 0.f, 0.f};
  v8f acc1 = {0.f, 0.f, 0.f, 0.f, 0.f, 0.f, 0.f, 0.f};

#pragma unroll
  for (int ky = 0; ky < 3; ++ky) {
#pragma unroll
    for (int kx = 0; kx < 3; ++kx) {
      const half_t* aa = wr + (ky * 3 + kx) * CPAD + kB;
      const half_t* bb = xn + ((size_t)(y + ky) * Ws + (px + kx)) * CPAD + kB;
#pragma unroll
      for (int c0 = 0; c0 < CPAD; c0 += 32) {
        v8h a0 = *(const v8h*)(aa + c0);
        v8h a1 = *(const v8h*)(aa + c0 + 8);
        v16h A = SHUF16(a0, a1);
        v8h b0 = *(const v8h*)(bb + c0);
        v8h b1 = *(const v8h*)(bb + c0 + 8);
        v16h B0 = SHUF16(b0, b1);
        acc0 = __builtin_amdgcn_wmma_f32_16x16x32_f16(false, A, false, B0,
                                                      (short)0, acc0, false, false);
        v8h b2 = *(const v8h*)(bb + 16 * CPAD + c0);
        v8h b3 = *(const v8h*)(bb + 16 * CPAD + c0 + 8);
        v16h B1 = SHUF16(b2, b3);
        acc1 = __builtin_amdgcn_wmma_f32_16x16x32_f16(false, A, false, B1,
                                                      (short)0, acc1, false, false);
      }
    }
  }

#pragma unroll
  for (int r = 0; r < 8; ++r) {
    int m = mt * 16 + hs * 8 + r;
    if (m < COUT) {
      float v0 = acc0[r] + bias[m];
      float v1 = acc1[r] + bias[m];
      if (LRELU) {
        v0 = (v0 >= 0.f) ? v0 : 0.2f * v0;
        v1 = (v1 >= 0.f) ? v1 : 0.2f * v1;
      }
      if (OUT16) {
        size_t i0, i1;
        if (OHALO) {
          size_t row = ((size_t)n * (H + 2) + (y + 1)) * Ws;
          i0 = (row + (px + 1))  * COUTP + m;
          i1 = (row + (px + 17)) * COUTP + m;
        } else {
          size_t row = ((size_t)n * H + y) * W;
          i0 = (row + px)        * COUTP + m;
          i1 = (row + px + 16)   * COUTP + m;
        }
        ((half_t*)outp)[i0] = (half_t)v0;
        ((half_t*)outp)[i1] = (half_t)v1;
      } else {
        size_t row = (((size_t)n * COUT + m) * H + y) * W;
        ((float*)outp)[row + px]      = v0;
        ((float*)outp)[row + px + 16] = v1;
      }
    }
  }
}

// ---------------------------------------------------------------------------
// 1x1 fuse conv (256->64) on WMMA; input pixel-major NHWC f16 [B][SL][256],
// output NHWC halo [B][(HL+2)][(WL+2)][64] + lrelu. 16x32 tile per wave.
// ---------------------------------------------------------------------------
__global__ void fuse_wmma(const half_t* __restrict__ x,
                          const half_t* __restrict__ w16,  // [64][256]
                          const float*  __restrict__ bias,
                          half_t*       __restrict__ outp) {
  constexpr int CT = 4;                          // 64/16
  constexpr int PTILES = SL_ / 32;
  int wave = (blockIdx.x * blockDim.x + threadIdx.x) >> 5;
  int lane = threadIdx.x & 31;
  if (wave >= B_ * PTILES * CT) return;

  int mt = wave % CT; int pt = wave / CT;
  int xt = pt % PTILES;
  int n  = pt / PTILES;

  const int hs  = lane >> 4;
  const int l15 = lane & 15;
  const int kA  = hs * 8;
  const int kB  = hs * 16;
  const int px  = xt * 32 + l15;                 // 32-tiles never straddle rows (224%32==0)

  const half_t* xb = x + ((size_t)n * SL_ + px) * 256 + kB;
  const half_t* wr = w16 + (size_t)(mt * 16 + l15) * 256 + kB;

  v8f acc0 = {0.f, 0.f, 0.f, 0.f, 0.f, 0.f, 0.f, 0.f};
  v8f acc1 = {0.f, 0.f, 0.f, 0.f, 0.f, 0.f, 0.f, 0.f};
#pragma unroll
  for (int c0 = 0; c0 < 256; c0 += 32) {
    v8h a0 = *(const v8h*)(wr + c0);
    v8h a1 = *(const v8h*)(wr + c0 + 8);
    v16h A = SHUF16(a0, a1);
    v8h b0 = *(const v8h*)(xb + c0);
    v8h b1 = *(const v8h*)(xb + c0 + 8);
    v16h B0 = SHUF16(b0, b1);
    acc0 = __builtin_amdgcn_wmma_f32_16x16x32_f16(false, A, false, B0,
                                                  (short)0, acc0, false, false);
    v8h b2 = *(const v8h*)(xb + 16 * 256 + c0);
    v8h b3 = *(const v8h*)(xb + 16 * 256 + c0 + 8);
    v16h B1 = SHUF16(b2, b3);
    acc1 = __builtin_amdgcn_wmma_f32_16x16x32_f16(false, A, false, B1,
                                                  (short)0, acc1, false, false);
  }

#pragma unroll
  for (int r = 0; r < 8; ++r) {
    int m = mt * 16 + hs * 8 + r;
    float v0 = acc0[r] + bias[m];
    float v1 = acc1[r] + bias[m];
    v0 = (v0 >= 0.f) ? v0 : 0.2f * v0;
    v1 = (v1 >= 0.f) ? v1 : 0.2f * v1;
    int p0 = px, p1 = px + 16;
    int y0 = p0 / WL, x0 = p0 % WL;
    int y1 = p1 / WL, x1 = p1 % WL;
    outp[(((size_t)n * (HL + 2) + (y0 + 1)) * (WL + 2) + (x0 + 1)) * 64 + m] = (half_t)v0;
    outp[(((size_t)n * (HL + 2) + (y1 + 1)) * (WL + 2) + (x1 + 1)) * 64 + m] = (half_t)v1;
  }
}

// ---------------------------------------------------------------------------
// ConvTranspose2d(k=3,s=2,p=1,op=1), NHWC in -> NHWC halo out (interior ptrs).
// ky = Y+1-2*iy (1 tap even Y, <=2 taps odd). Repacked weights [9][Cout][CINP]
// make both streams channel-contiguous => vectorized loads. f32 accumulate.
// ---------------------------------------------------------------------------
template<int CINP>
__global__ void convt_nhwc(const half_t* __restrict__ xin,  // interior base, ch-stride CINP
                           const float*  __restrict__ wt,   // [9][Cout][CINP]
                           const float*  __restrict__ bias,
                           half_t*       __restrict__ out,  // interior base, ch-stride OCS
                           int N, int Cout, int Hin, int Win,
                           int inRowW, long long inN,
                           int outRowW, int OCS, long long outN) {
  const int Hout = Hin * 2, Wout = Win * 2;
  long long i = (long long)blockIdx.x * blockDim.x + threadIdx.x;
  long long total = (long long)N * Hout * Wout * Cout;
  if (i >= total) return;
  int o = (int)(i % Cout); long long t = i / Cout;
  int X = (int)(t % Wout); t /= Wout;
  int Y = (int)(t % Hout);
  int n = (int)(t / Hout);

  int iyv[2], kyv[2], nky;
  if (Y & 1) {
    nky = 0;
    iyv[nky] = (Y - 1) >> 1; kyv[nky] = 2; ++nky;
    if (((Y + 1) >> 1) < Hin) { iyv[nky] = (Y + 1) >> 1; kyv[nky] = 0; ++nky; }
  } else { iyv[0] = Y >> 1; kyv[0] = 1; nky = 1; }
  int ixv[2], kxv[2], nkx;
  if (X & 1) {
    nkx = 0;
    ixv[nkx] = (X - 1) >> 1; kxv[nkx] = 2; ++nkx;
    if (((X + 1) >> 1) < Win) { ixv[nkx] = (X + 1) >> 1; kxv[nkx] = 0; ++nkx; }
  } else { ixv[0] = X >> 1; kxv[0] = 1; nkx = 1; }

  const half_t* xb = xin + (size_t)n * inN;
  float acc = bias[o];
  for (int a = 0; a < nky; ++a) {
    for (int b2 = 0; b2 < nkx; ++b2) {
      const half_t* xr = xb + ((size_t)iyv[a] * inRowW + ixv[b2]) * CINP;
      const float*  wr = wt + ((size_t)(kyv[a] * 3 + kxv[b2]) * Cout + o) * CINP;
      float s = 0.f;
#pragma unroll
      for (int c = 0; c < CINP; c += 8) {
        v8h xv = *(const v8h*)(xr + c);
        v8f wv = *(const v8f*)(wr + c);
#pragma unroll
        for (int j = 0; j < 8; ++j) s += (float)xv[j] * wv[j];
      }
      acc += s;
    }
  }
  out[(size_t)n * outN + ((size_t)Y * outRowW + X) * OCS + o] = (half_t)acc;
}

// ---------------------------------------------------------------------------
// Channel softmax over [Bn, Cch, S] f32 (stride-S channels); safe in place.
// ---------------------------------------------------------------------------
__global__ void softmax_ch(const float* __restrict__ in, float* __restrict__ out,
                           int Bn, int Cch, int S) {
  int i = blockIdx.x * blockDim.x + threadIdx.x;
  if (i >= Bn * S) return;
  int b = i / S, p = i % S;
  const float* ip = in  + (size_t)b * Cch * S + p;
  float*       op = out + (size_t)b * Cch * S + p;
  float mx = -3.0e38f;
  for (int c = 0; c < Cch; ++c) mx = fmaxf(mx, ip[(size_t)c * S]);
  float sum = 0.f;
  for (int c = 0; c < Cch; ++c) sum += __expf(ip[(size_t)c * S] - mx);
  float inv = 1.f / sum;
  for (int c = 0; c < Cch; ++c) op[(size_t)c * S] = __expf(ip[(size_t)c * S] - mx) * inv;
}

// ---------------------------------------------------------------------------
// AdaCoF: 25-tap deformable bilinear gather with replication padding (pad=2),
// occlusion-weighted, accumulated into out for frame t.
// ---------------------------------------------------------------------------
__global__ void adacof_accum(const float* __restrict__ frame,
                             const float* __restrict__ occl, int t,
                             const float* __restrict__ wgt,
                             const float* __restrict__ alp,
                             const float* __restrict__ bet,
                             float* __restrict__ out) {
  const int S = SH_;
  int i = blockIdx.x * blockDim.x + threadIdx.x;
  if (i >= B_ * S) return;
  int b = i / S, p = i % S;
  int y = p / WH, x = p % WH;
  const int Hp = HH + 4, Wp = WH + 4;
  float s0 = 0.f, s1 = 0.f, s2 = 0.f;

  for (int tap = 0; tap < F2_; ++tap) {
    int py = tap / 5, qx = tap % 5;
    size_t o = ((size_t)b * F2_ + tap) * S + p;
    float aa = alp[o], bb = bet[o], ww = wgt[o];
    float yf = (float)(y + py) + aa;
    float xf = (float)(x + qx) + bb;
    float y0f = floorf(yf), x0f = floorf(xf);
    float wy = yf - y0f, wx = xf - x0f;
    int y0p = imin(imax((int)y0f, 0), Hp - 1);
    int y1p = imin(y0p + 1, Hp - 1);
    int x0p = imin(imax((int)x0f, 0), Wp - 1);
    int x1p = imin(x0p + 1, Wp - 1);
    int y0 = imin(imax(y0p - 2, 0), HH - 1);
    int y1 = imin(imax(y1p - 2, 0), HH - 1);
    int x0 = imin(imax(x0p - 2, 0), WH - 1);
    int x1 = imin(imax(x1p - 2, 0), WH - 1);
    float w00 = (1.f - wy) * (1.f - wx), w01 = (1.f - wy) * wx;
    float w10 = wy * (1.f - wx),         w11 = wy * wx;
    size_t i00 = (size_t)y0 * WH + x0, i01 = (size_t)y0 * WH + x1;
    size_t i10 = (size_t)y1 * WH + x0, i11 = (size_t)y1 * WH + x1;
    const float* img0 = frame + ((size_t)b * 3 + 0) * S;
    const float* img1 = frame + ((size_t)b * 3 + 1) * S;
    const float* img2 = frame + ((size_t)b * 3 + 2) * S;
    s0 += ww * (w00 * img0[i00] + w01 * img0[i01] + w10 * img0[i10] + w11 * img0[i11]);
    s1 += ww * (w00 * img1[i00] + w01 * img1[i01] + w10 * img1[i10] + w11 * img1[i11]);
    s2 += ww * (w00 * img2[i00] + w01 * img2[i01] + w10 * img2[i10] + w11 * img2[i11]);
  }
  float ov = occl[((size_t)b * T_ + t) * S + p];
  out[((size_t)b * 3 + 0) * S + p] += ov * s0;
  out[((size_t)b * 3 + 1) * S + p] += ov * s1;
  out[((size_t)b * 3 + 2) * S + p] += ov * s2;
}

// ---------------------------------------------------------------------------
// Host orchestration
// ---------------------------------------------------------------------------
static inline int grid1(long long n, int blk) { return (int)((n + blk - 1) / blk); }

extern "C" void kernel_launch(void* const* d_in, const int* in_sizes, int n_in,
                              void* d_out, int out_size, void* d_ws, size_t ws_size,
                              hipStream_t stream) {
  (void)in_sizes; (void)n_in; (void)ws_size;

  const float* fea    = (const float*)d_in[0];
  const float* frames = (const float*)d_in[1];
  const float* fuse_w = (const float*)d_in[2];
  const float* fuse_b = (const float*)d_in[3];
  const float *sn_c1w[3], *sn_c1b[3], *sn_c2w[3], *sn_c2b[3];
  const float *sn_ctw[3], *sn_ctb[3], *sn_c3w[3], *sn_c3b[3];
  for (int s = 0; s < 3; ++s) {                  // s=0 weight, 1 alpha, 2 beta
    int k = 4 + 8 * s;
    sn_c1w[s] = (const float*)d_in[k + 0]; sn_c1b[s] = (const float*)d_in[k + 1];
    sn_c2w[s] = (const float*)d_in[k + 2]; sn_c2b[s] = (const float*)d_in[k + 3];
    sn_ctw[s] = (const float*)d_in[k + 4]; sn_ctb[s] = (const float*)d_in[k + 5];
    sn_c3w[s] = (const float*)d_in[k + 6]; sn_c3b[s] = (const float*)d_in[k + 7];
  }
  const float* occ_c1w = (const float*)d_in[28]; const float* occ_c1b = (const float*)d_in[29];
  const float* occ_c2w = (const float*)d_in[30]; const float* occ_c2b = (const float*)d_in[31];
  const float* occ_ctw = (const float*)d_in[32]; const float* occ_ctb = (const float*)d_in[33];
  const float* occ_c3w = (const float*)d_in[34]; const float* occ_c3b = (const float*)d_in[35];

  char* wsb = (char*)d_ws;
  size_t off = 0;
  auto alloc = [&](size_t bytes) -> char* {
    char* p = wsb + off;
    off = (off + bytes + 255) & ~(size_t)255;
    return p;
  };

  // repacked WMMA weights (tap-major, padded)
  half_t* w16_fuse = (half_t*)alloc((size_t)64 * 256 * 2);        // [64][256]
  half_t* w16_occ1 = (half_t*)alloc((size_t)64 * 576 * 2);
  half_t* w16_occ2 = (half_t*)alloc((size_t)64 * 576 * 2);
  half_t* w16_occ3 = (half_t*)alloc((size_t)16 * 576 * 2);
  half_t *w16_c1[3], *w16_c2[3], *w16_c3[3];
  for (int s = 0; s < 3; ++s) {
    w16_c1[s] = (half_t*)alloc((size_t)64 * 576 * 2);
    w16_c2[s] = (half_t*)alloc((size_t)32 * 576 * 2);
    w16_c3[s] = (half_t*)alloc((size_t)32 * 288 * 2);
  }
  // repacked convT weights [9][Cout][CinP] f32
  float* wt_occ = (float*)alloc((size_t)9 * 64 * 64 * 4);
  float* wt_sn[3];
  for (int s = 0; s < 3; ++s) wt_sn[s] = (float*)alloc((size_t)9 * 25 * 32 * 4);

  // activations
  const size_t N_FEAT  = (size_t)B_ * SL_ * 256;                  // pixel-major fuse input
  const size_t N_PAD64 = (size_t)B_ * (HL + 2) * (WL + 2) * 64;   // NHWC halo low-res
  const size_t N_S1    = (size_t)B_ * HL * WL * 32;               // NHWC flat, 25->pad32
  const size_t N_S2    = (size_t)B_ * (HH + 2) * (WH + 2) * 32;   // NHWC halo hi-res, pad32
  const size_t N_XB2   = (size_t)B_ * (HH + 2) * (WH + 2) * 64;   // NHWC halo hi-res, 64ch
  half_t* feaT   = (half_t*)alloc(N_FEAT * 2);
  half_t* pa16   = (half_t*)alloc(N_PAD64 * 2);
  half_t* pb16   = (half_t*)alloc(N_PAD64 * 2);
  half_t* feat16 = (half_t*)alloc(N_PAD64 * 2);
  half_t* s1_16  = (half_t*)alloc(N_S1 * 2);
  half_t* s2pad  = (half_t*)alloc(N_S2 * 2);
  half_t* xbig2  = (half_t*)alloc(N_XB2 * 2);
  float* occ_lg  = (float*)alloc((size_t)B_ * T_ * SH_ * 4);
  float* occl    = (float*)alloc((size_t)B_ * T_ * SH_ * 4);
  float* head_w  = (float*)alloc((size_t)B_ * F2_ * SH_ * 4);
  float* head_a  = (float*)alloc((size_t)B_ * F2_ * SH_ * 4);
  float* head_b  = (float*)alloc((size_t)B_ * F2_ * SH_ * 4);
  float* heads[3] = {head_w, head_a, head_b};

  const int BLK = 256;

  // ---- zero halo'd / padded f16 buffers ----
  auto zero16 = [&](half_t* p, size_t nh) {
    long long nw = (long long)(nh / 2);
    zero_u32<<<grid1(nw, BLK), BLK, 0, stream>>>((uint32_t*)p, nw);
  };
  zero16(pa16, N_PAD64); zero16(pb16, N_PAD64); zero16(feat16, N_PAD64);
  zero16(s1_16, N_S1);   zero16(s2pad, N_S2);   zero16(xbig2, N_XB2);

  // ---- repack weights ----
  auto rp = [&](const float* w, half_t* o, int Co, int Ci, int CoP, int CiP, int taps) {
    int n = CoP * taps * CiP;
    repack_w<<<grid1(n, BLK), BLK, 0, stream>>>(w, o, Co, Ci, CoP, CiP, taps);
  };
  rp(fuse_w, w16_fuse, 64, 256, 64, 256, 1);
  rp(occ_c1w, w16_occ1, 64, 64, 64, 64, 9);
  rp(occ_c2w, w16_occ2, 64, 64, 64, 64, 9);
  rp(occ_c3w, w16_occ3, 4, 64, 16, 64, 9);
  for (int s = 0; s < 3; ++s) {
    rp(sn_c1w[s], w16_c1[s], 64, 64, 64, 64, 9);
    rp(sn_c2w[s], w16_c2[s], 25, 64, 32, 64, 9);
    rp(sn_c3w[s], w16_c3[s], 25, 25, 32, 32, 9);
  }
  repack_wt<<<grid1(9 * 64 * 64, BLK), BLK, 0, stream>>>(occ_ctw, wt_occ, 64, 64, 64);
  for (int s = 0; s < 3; ++s)
    repack_wt<<<grid1(9 * 25 * 32, BLK), BLK, 0, stream>>>(sn_ctw[s], wt_sn[s], 25, 25, 32);

  // ---- occlusion branch ----
  fuse_tr<<<B_ * (SL_ / 64) * 4, BLK, 0, stream>>>(fea, feaT);
  {  // fuse 1x1 256->64 + lrelu -> pa16 (NHWC halo)
    long long wv = (long long)B_ * (SL_ / 32) * 4;
    fuse_wmma<<<grid1(wv * 32, BLK), BLK, 0, stream>>>(feaT, w16_fuse, fuse_b, pa16);
  }
  {  // occ c1: 64->64 +lrelu, pa16 -> pb16
    long long wv = (long long)B_ * HL * (WL / 32) * 4;
    conv3_wmma<HL, WL, 64, 64, 64, true, true, true>
        <<<grid1(wv * 32, BLK), BLK, 0, stream>>>(pa16, w16_occ1, occ_c1b, pb16, B_);
  }
  {  // occ c2: 64->64 +lrelu, pb16 -> pa16
    long long wv = (long long)B_ * HL * (WL / 32) * 4;
    conv3_wmma<HL, WL, 64, 64, 64, true, true, true>
        <<<grid1(wv * 32, BLK), BLK, 0, stream>>>(pb16, w16_occ2, occ_c2b, pa16, B_);
  }
  {  // occ convT 64->64 x2 -> xbig2 (NHWC halo)
    long long n = (long long)B_ * 64 * SH_;
    convt_nhwc<64><<<grid1(n, BLK), BLK, 0, stream>>>(
        pa16 + ((size_t)(WL + 2) + 1) * 64, wt_occ, occ_ctb,
        xbig2 + ((size_t)(WH + 2) + 1) * 64,
        B_, 64, HL, WL,
        WL + 2, (long long)(HL + 2) * (WL + 2) * 64,
        WH + 2, 64, (long long)(HH + 2) * (WH + 2) * 64);
  }
  {  // occ c3: 64->4, f32 NCHW logits
    long long wv = (long long)B_ * HH * (WH / 32) * 1;
    conv3_wmma<HH, WH, 64, 16, 4, false, false, false>
        <<<grid1(wv * 32, BLK), BLK, 0, stream>>>(xbig2, w16_occ3, occ_c3b, occ_lg, B_);
  }
  softmax_ch<<<grid1((long long)B_ * SH_, BLK), BLK, 0, stream>>>(occ_lg, occl, B_, T_, SH_);

  // ---- zero output accumulator ----
  zero_f32<<<grid1(out_size, BLK), BLK, 0, stream>>>((float*)d_out, out_size);

  // ---- per-frame subnets + AdaCoF ----
  for (int t = 0; t < T_; ++t) {
    feat_tr<<<B_ * (SL_ / 64), BLK, 0, stream>>>(fea, feat16, t);
    for (int s = 0; s < 3; ++s) {
      {  // c1: 64->64 +lrelu, feat16 -> pa16
        long long wv = (long long)B_ * HL * (WL / 32) * 4;
        conv3_wmma<HL, WL, 64, 64, 64, true, true, true>
            <<<grid1(wv * 32, BLK), BLK, 0, stream>>>(feat16, w16_c1[s], sn_c1b[s], pa16, B_);
      }
      {  // c2: 64->25 +lrelu, pa16 -> s1_16 (NHWC flat, 32ch stride)
        long long wv = (long long)B_ * HL * (WL / 32) * 2;
        conv3_wmma<HL, WL, 64, 32, 25, true, true, false>
            <<<grid1(wv * 32, BLK), BLK, 0, stream>>>(pa16, w16_c2[s], sn_c2b[s], s1_16, B_);
      }
      {  // convT 25->25 x2 -> s2pad (NHWC halo, 32ch stride)
        long long n = (long long)B_ * F2_ * SH_;
        convt_nhwc<32><<<grid1(n, BLK), BLK, 0, stream>>>(
            s1_16, wt_sn[s], sn_ctb[s],
            s2pad + ((size_t)(WH + 2) + 1) * 32,
            B_, F2_, HL, WL,
            WL, (long long)HL * WL * 32,
            WH + 2, 32, (long long)(HH + 2) * (WH + 2) * 32);
      }
      {  // c3: 25->25, f32 NCHW head
        long long wv = (long long)B_ * HH * (WH / 32) * 2;
        conv3_wmma<HH, WH, 32, 32, 25, false, false, false>
            <<<grid1(wv * 32, BLK), BLK, 0, stream>>>(s2pad, w16_c3[s], sn_c3b[s], heads[s], B_);
      }
    }
    softmax_ch<<<grid1((long long)B_ * SH_, BLK), BLK, 0, stream>>>(head_w, head_w, B_, F2_, SH_);
    adacof_accum<<<grid1((long long)B_ * SH_, BLK), BLK, 0, stream>>>(
        frames + (size_t)t * B_ * 3 * SH_, occl, t, head_w, head_a, head_b, (float*)d_out);
  }
}
